// CustomMultiheadAttention_18133351924190
// MI455X (gfx1250) — compile-verified, matrix operand order checked by reading, not run on hardware
//
#include <hip/hip_runtime.h>
#include <math.h>

// ---------------------------------------------------------------------------
// Problem constants (from the reference): B=4, S=4096, E=1024, H=16, R=512, D=64
// ---------------------------------------------------------------------------
#define B_   4
#define S_   4096
#define E_   1024
#define H_   16
#define R_   512
#define D_   64
#define BS_  (B_ * S_)

// ---------------------------------------------------------------------------
// Vector types (avoid HIP struct vector types inside unions)
// ---------------------------------------------------------------------------
typedef __attribute__((ext_vector_type(16))) __bf16 v16bf;
typedef __attribute__((ext_vector_type(8)))  float  v8f;
typedef __attribute__((ext_vector_type(4)))  int    i4;
typedef __attribute__((ext_vector_type(4)))  float  f4;

union FragBF {
    v16bf v;
    i4    q[2];
};

// ---------------------------------------------------------------------------
// f32 -> bf16 conversion (4 elements / thread); all sizes are multiples of 4.
// ---------------------------------------------------------------------------
__global__ __launch_bounds__(256) void cvt_f32_bf16(const float* __restrict__ x,
                                                    __bf16* __restrict__ y, int n) {
    int i = (blockIdx.x * 256 + threadIdx.x) * 4;
    if (i + 3 < n) {
        f4 v = *(const f4*)(x + i);
        y[i + 0] = (__bf16)v.x;
        y[i + 1] = (__bf16)v.y;
        y[i + 2] = (__bf16)v.z;
        y[i + 3] = (__bf16)v.w;
    }
}

// ---------------------------------------------------------------------------
// Generic NT GEMM: C[M,N] = A[M,K] * B[N,K]^T   (bf16 in, f32 accumulate)
//   BM x BN block tile, BK = 32, 256 threads = 8 waves arranged WRM x WRN.
//   Each wave computes (BM/WRM) x (BN/WRN) via 16x16x32 bf16 WMMA.
//   LDS double-buffered: one workgroup barrier per K-step; global loads for
//   tile i+1 are issued before the WMMAs of tile i (latency overlap).
// Epilogues (EPI):
//   0: store bf16, row-major [M, ldc]
//   1: +bias[n], store f32 row-major            (final out projection)
//   2: +bias[n], elu(x)+1, store bf16 head-major [B,H,S,D]   (q/k up-proj)
//   3: +bias[n],           store bf16 head-major [B,H,S,D]   (v up-proj)
//   4: store f32, batched (blockIdx.z), row-major [M, ldc]   (num GEMM)
// ---------------------------------------------------------------------------
template <int BM, int BN, int WRM, int WRN, int EPI>
__global__ __launch_bounds__(256) void gemm_nt(
    const __bf16* __restrict__ A, const __bf16* __restrict__ Bm,
    const float* __restrict__ bias, void* __restrict__ Cv,
    int M, int N, int K, int lda, int ldb, int ldc,
    long long sAb, long long sBb, long long sCb) {

    constexpr int BK   = 32;
    constexpr int STR  = 40;                // padded LDS row stride (halves): 80B, 16B-aligned
    constexpr int WM   = BM / WRM;
    constexpr int WN   = BN / WRN;
    constexpr int MI   = WM / 16;
    constexpr int NI   = WN / 16;
    constexpr int PERA = BM * BK / 256;     // halves staged per thread (A)
    constexpr int PERB = BN * BK / 256;     // halves staged per thread (B)
    constexpr int NRA  = PERA / 8;          // 16B vectors per thread (A)
    constexpr int NRB  = PERB / 8;          // 16B vectors per thread (B)

    __shared__ alignas(16) __bf16 As[2][BM * STR];
    __shared__ alignas(16) __bf16 Bs[2][BN * STR];

    A  += (size_t)blockIdx.z * (size_t)sAb;
    Bm += (size_t)blockIdx.z * (size_t)sBb;

    const int t    = threadIdx.x;
    const int lane = t & 31;
    const int wave = t >> 5;
    const int mW   = (wave / WRN) * WM;
    const int nW   = (wave % WRN) * WN;

    const size_t aRow0 = (size_t)blockIdx.y * BM;
    const size_t bRow0 = (size_t)blockIdx.x * BN;

    const v8f vzero = {0.f, 0.f, 0.f, 0.f, 0.f, 0.f, 0.f, 0.f};
    v8f acc[MI][NI];
#pragma unroll
    for (int mi = 0; mi < MI; ++mi)
#pragma unroll
        for (int ni = 0; ni < NI; ++ni) acc[mi][ni] = vzero;

    // staging registers (global -> reg -> LDS)
    i4 ra[NRA], rb[NRB];
    auto load_regs = [&](int k0) {
#pragma unroll
        for (int i = 0; i < NRA; ++i) {
            int L = t * PERA + i * 8;
            ra[i] = *(const i4*)(A + (aRow0 + (L >> 5)) * (size_t)lda + k0 + (L & 31));
        }
#pragma unroll
        for (int i = 0; i < NRB; ++i) {
            int L = t * PERB + i * 8;
            rb[i] = *(const i4*)(Bm + (bRow0 + (L >> 5)) * (size_t)ldb + k0 + (L & 31));
        }
    };
    auto store_lds = [&](int buf) {
#pragma unroll
        for (int i = 0; i < NRA; ++i) {
            int L = t * PERA + i * 8;
            *(i4*)(&As[buf][(L >> 5) * STR + (L & 31)]) = ra[i];
        }
#pragma unroll
        for (int i = 0; i < NRB; ++i) {
            int L = t * PERB + i * 8;
            *(i4*)(&Bs[buf][(L >> 5) * STR + (L & 31)]) = rb[i];
        }
    };

    // 16-bit fragment K-chunk select per the CDNA5 16x32 A/B lane layout:
    // lanes 0-15 hold K chunks {0..7, 16..23}; lanes 16-31 hold {8..15, 24..31}.
    const int c0 = (lane & 16) ? 8 : 0;

    // prologue: stage tile 0
    load_regs(0);
    store_lds(0);
    __syncthreads();

    const int nsteps = K / BK;
    for (int step = 0; step < nsteps; ++step) {
        const int cur = step & 1;
        const bool more = (step + 1) < nsteps;
        if (more) load_regs((step + 1) * BK);   // overlap with WMMAs below

        // ---- load fragments from LDS (current buffer) ----
        FragBF a[MI], b[NI];
#pragma unroll
        for (int mi = 0; mi < MI; ++mi) {
            const __bf16* p = &As[cur][(mW + mi * 16 + (lane & 15)) * STR];
            a[mi].q[0] = *(const i4*)(p + c0);
            a[mi].q[1] = *(const i4*)(p + c0 + 16);
        }
#pragma unroll
        for (int ni = 0; ni < NI; ++ni) {
            const __bf16* p = &Bs[cur][(nW + ni * 16 + (lane & 15)) * STR];
            b[ni].q[0] = *(const i4*)(p + c0);
            b[ni].q[1] = *(const i4*)(p + c0 + 16);
        }

        // ---- WMMA: D = A x B + C (f32 accumulate) ----
#pragma unroll
        for (int mi = 0; mi < MI; ++mi)
#pragma unroll
            for (int ni = 0; ni < NI; ++ni)
                acc[mi][ni] = __builtin_amdgcn_wmma_f32_16x16x32_bf16(
                    false, a[mi].v, false, b[ni].v, (short)0, acc[mi][ni],
                    false, false);

        if (more) {
            store_lds(cur ^ 1);   // write-side of double buffer (no conflict with cur)
            __syncthreads();      // single barrier per K-step
        }
    }

    // ---- epilogue: C layout = 8 VGPRs; lanes 0-15 -> M=j, lanes 16-31 -> M=j+8
#pragma unroll
    for (int mi = 0; mi < MI; ++mi) {
#pragma unroll
        for (int ni = 0; ni < NI; ++ni) {
            const int n0 = blockIdx.x * BN + nW + ni * 16 + (lane & 15);
            const int m0 = blockIdx.y * BM + mW + mi * 16 + ((lane >> 4) << 3);
            float bval = 0.f;
            if constexpr (EPI == 1 || EPI == 2 || EPI == 3) bval = bias[n0];
#pragma unroll
            for (int j = 0; j < 8; ++j) {
                float v   = acc[mi][ni][j];
                int  grow = m0 + j;
                if constexpr (EPI == 0) {
                    ((__bf16*)Cv)[(size_t)grow * ldc + n0] = (__bf16)v;
                } else if constexpr (EPI == 1) {
                    ((float*)Cv)[(size_t)grow * ldc + n0] = v + bval;
                } else if constexpr (EPI == 2 || EPI == 3) {
                    v += bval;
                    if constexpr (EPI == 2) v = (v > 0.f) ? (v + 1.f) : __expf(v);
                    // [BS, E] -> head-major [B, H, S, D]
                    int bb = grow >> 12, nseq = grow & (S_ - 1);
                    int hh = n0 >> 6,   dd   = n0 & (D_ - 1);
                    ((__bf16*)Cv)[(((size_t)(bb * H_ + hh) * S_) + nseq) * D_ + dd] =
                        (__bf16)v;
                } else {  // EPI == 4 (batched f32)
                    ((float*)Cv)[(size_t)blockIdx.z * (size_t)sCb +
                                 (size_t)grow * ldc + n0] = v;
                }
            }
        }
    }
}

// ---------------------------------------------------------------------------
// k_sum[b,h,d] = sum_n kfeat[b,h,n,d]      (deterministic, no atomics)
// grid = B*H blocks, 256 threads: 4-way S split + LDS reduce
// ---------------------------------------------------------------------------
__global__ __launch_bounds__(256) void ksum_kernel(const __bf16* __restrict__ kf,
                                                   float* __restrict__ ksum) {
    const int bh  = blockIdx.x;
    const int d   = threadIdx.x & 63;
    const int sub = threadIdx.x >> 6;
    const __bf16* base = kf + ((size_t)bh * S_ + (size_t)sub * (S_ / 4)) * D_ + d;
    float acc = 0.f;
    for (int n = 0; n < S_ / 4; ++n) acc += (float)base[(size_t)n * D_];
    __shared__ float red[256];
    red[threadIdx.x] = acc;
    __syncthreads();
    if (sub == 0)
        ksum[bh * D_ + d] = red[d] + red[d + 64] + red[d + 128] + red[d + 192];
}

// ---------------------------------------------------------------------------
// kvT[b,h,e,d] = sum_n kfeat[b,h,n,d] * vfeat[b,h,n,e]
// (stored transposed so num = q @ kv becomes a standard NT GEMM)
// grid = B*H blocks, 256 threads; 16-row LDS staging; thread owns (d, 16 e's)
// ---------------------------------------------------------------------------
__global__ __launch_bounds__(256) void kv_kernel(const __bf16* __restrict__ kf,
                                                 const __bf16* __restrict__ vf,
                                                 float* __restrict__ kvT) {
    __shared__ __bf16 ks[16 * D_];
    __shared__ __bf16 vs[16 * D_];
    const int bh = blockIdx.x;
    const int d  = threadIdx.x & 63;
    const int eg = threadIdx.x >> 6;  // 0..3 -> e range [eg*16, eg*16+16)
    const __bf16* kb = kf + (size_t)bh * S_ * D_;
    const __bf16* vb = vf + (size_t)bh * S_ * D_;
    float acc[16];
#pragma unroll
    for (int e = 0; e < 16; ++e) acc[e] = 0.f;

    for (int n0 = 0; n0 < S_; n0 += 16) {
        __syncthreads();
        const int r = threadIdx.x >> 4;
        const int c = (threadIdx.x & 15) * 4;
#pragma unroll
        for (int j = 0; j < 4; ++j) {
            ks[r * D_ + c + j] = kb[(size_t)(n0 + r) * D_ + c + j];
            vs[r * D_ + c + j] = vb[(size_t)(n0 + r) * D_ + c + j];
        }
        __syncthreads();
#pragma unroll
        for (int r2 = 0; r2 < 16; ++r2) {
            const float kd = (float)ks[r2 * D_ + d];
#pragma unroll
            for (int e = 0; e < 16; ++e)
                acc[e] += kd * (float)vs[r2 * D_ + eg * 16 + e];
        }
    }
#pragma unroll
    for (int e = 0; e < 16; ++e)
        kvT[(size_t)bh * (D_ * D_) + (size_t)(eg * 16 + e) * D_ + d] = acc[e];
}

// ---------------------------------------------------------------------------
// denom[b,h,n] = dot(qfeat[b,h,n,:], ksum[b,h,:])   (one thread per row)
// ---------------------------------------------------------------------------
__global__ __launch_bounds__(256) void denom_kernel(const __bf16* __restrict__ qf,
                                                    const float* __restrict__ ks,
                                                    float* __restrict__ denom) {
    const size_t i  = (size_t)blockIdx.x * 256 + threadIdx.x;  // over B*H*S
    const int    bh = (int)(i >> 12);
    const __bf16* row = qf + i * D_;
    const float*  kk  = ks + bh * D_;
    float acc = 0.f;
#pragma unroll 8
    for (int dd = 0; dd < D_; ++dd) acc += (float)row[dd] * kk[dd];
    denom[i] = acc;
}

// ---------------------------------------------------------------------------
// attn[b, n, h*D+d] = num[b,h,n,d] / (denom[b,h,n] + 1e-6)  (bf16 out)
// ---------------------------------------------------------------------------
__global__ __launch_bounds__(256) void combine_kernel(const float* __restrict__ num,
                                                      const float* __restrict__ denom,
                                                      __bf16* __restrict__ attn) {
    const size_t i   = (size_t)blockIdx.x * 256 + threadIdx.x;  // over B*H*S*D
    const int    dd  = (int)(i & (D_ - 1));
    const size_t row = i >> 6;                                   // bh*S + n
    const int    ns  = (int)(row & (S_ - 1));
    const int    bh  = (int)(row >> 12);
    const int    bb  = bh >> 4, hh = bh & (H_ - 1);
    const float  v   = num[i] / (denom[row] + 1e-6f);
    attn[((size_t)(bb * S_ + ns)) * E_ + hh * D_ + dd] = (__bf16)v;
}

// ---------------------------------------------------------------------------
// Host-side orchestration
// ---------------------------------------------------------------------------
extern "C" void kernel_launch(void* const* d_in, const int* in_sizes, int n_in,
                              void* d_out, int out_size, void* d_ws, size_t ws_size,
                              hipStream_t stream) {
    const float* q_in  = (const float*)d_in[0];
    const float* k_in  = (const float*)d_in[1];
    const float* v_in  = (const float*)d_in[2];
    const float* qd_w  = (const float*)d_in[3];
    const float* qu_w  = (const float*)d_in[4];
    const float* qu_b  = (const float*)d_in[5];
    const float* kd_w  = (const float*)d_in[6];
    const float* ku_w  = (const float*)d_in[7];
    const float* ku_b  = (const float*)d_in[8];
    const float* vd_w  = (const float*)d_in[9];
    const float* vu_w  = (const float*)d_in[10];
    const float* vu_b  = (const float*)d_in[11];
    const float* out_w = (const float*)d_in[12];
    const float* out_b = (const float*)d_in[13];
    float* out = (float*)d_out;

    // workspace carve-out (256B aligned)
    char* p = (char*)d_ws;
    auto take = [&](size_t bytes) -> char* {
        char* r = p;
        p += (bytes + 255) & ~(size_t)255;
        return r;
    };
    __bf16* xq   = (__bf16*)take((size_t)BS_ * E_ * 2);   // later reused as qfeat [B,H,S,D]
    __bf16* xk   = (__bf16*)take((size_t)BS_ * E_ * 2);   // later reused as kfeat
    __bf16* xv   = (__bf16*)take((size_t)BS_ * E_ * 2);   // later reused as vfeat
    __bf16* wdq  = (__bf16*)take((size_t)R_ * E_ * 2);
    __bf16* wdk  = (__bf16*)take((size_t)R_ * E_ * 2);
    __bf16* wdv  = (__bf16*)take((size_t)R_ * E_ * 2);
    __bf16* wuq  = (__bf16*)take((size_t)E_ * R_ * 2);
    __bf16* wuk  = (__bf16*)take((size_t)E_ * R_ * 2);
    __bf16* wuv  = (__bf16*)take((size_t)E_ * R_ * 2);
    __bf16* wout = (__bf16*)take((size_t)E_ * E_ * 2);
    __bf16* Tq   = (__bf16*)take((size_t)BS_ * R_ * 2);
    __bf16* Tk   = (__bf16*)take((size_t)BS_ * R_ * 2);
    __bf16* Tv   = (__bf16*)take((size_t)BS_ * R_ * 2);
    float*  kvTf = (float*) take((size_t)B_ * H_ * D_ * D_ * 4);
    __bf16* kvTb = (__bf16*)take((size_t)B_ * H_ * D_ * D_ * 2);
    float*  ksum = (float*) take((size_t)B_ * H_ * D_ * 4);
    float*  deno = (float*) take((size_t)B_ * H_ * S_ * 4);
    float*  numb = (float*) take((size_t)B_ * H_ * S_ * D_ * 4);
    __bf16* attn = (__bf16*)take((size_t)BS_ * E_ * 2);

    const dim3 blk(256);

    // 1) fp32 -> bf16 conversions (inputs + weights)
    cvt_f32_bf16<<<(BS_ * E_) / 1024, blk, 0, stream>>>(q_in, xq, BS_ * E_);
    cvt_f32_bf16<<<(BS_ * E_) / 1024, blk, 0, stream>>>(k_in, xk, BS_ * E_);
    cvt_f32_bf16<<<(BS_ * E_) / 1024, blk, 0, stream>>>(v_in, xv, BS_ * E_);
    cvt_f32_bf16<<<(R_ * E_) / 1024, blk, 0, stream>>>(qd_w, wdq, R_ * E_);
    cvt_f32_bf16<<<(R_ * E_) / 1024, blk, 0, stream>>>(kd_w, wdk, R_ * E_);
    cvt_f32_bf16<<<(R_ * E_) / 1024, blk, 0, stream>>>(vd_w, wdv, R_ * E_);
    cvt_f32_bf16<<<(E_ * R_) / 1024, blk, 0, stream>>>(qu_w, wuq, E_ * R_);
    cvt_f32_bf16<<<(E_ * R_) / 1024, blk, 0, stream>>>(ku_w, wuk, E_ * R_);
    cvt_f32_bf16<<<(E_ * R_) / 1024, blk, 0, stream>>>(vu_w, wuv, E_ * R_);
    cvt_f32_bf16<<<(E_ * E_) / 1024, blk, 0, stream>>>(out_w, wout, E_ * E_);

    // 2) down projections: T = X @ Wd^T   [BS,R], bf16 out
    gemm_nt<128, 256, 2, 4, 0><<<dim3(R_ / 256, BS_ / 128, 1), blk, 0, stream>>>(
        xq, wdq, nullptr, Tq, BS_, R_, E_, E_, E_, R_, 0, 0, 0);
    gemm_nt<128, 256, 2, 4, 0><<<dim3(R_ / 256, BS_ / 128, 1), blk, 0, stream>>>(
        xk, wdk, nullptr, Tk, BS_, R_, E_, E_, E_, R_, 0, 0, 0);
    gemm_nt<128, 256, 2, 4, 0><<<dim3(R_ / 256, BS_ / 128, 1), blk, 0, stream>>>(
        xv, wdv, nullptr, Tv, BS_, R_, E_, E_, E_, R_, 0, 0, 0);

    // 3) up projections with fused bias (+elu+1 for q/k) and head-major bf16 store
    //    (feature maps overwrite the now-dead bf16 input copies)
    gemm_nt<128, 256, 2, 4, 2><<<dim3(E_ / 256, BS_ / 128, 1), blk, 0, stream>>>(
        Tq, wuq, qu_b, xq, BS_, E_, R_, R_, R_, E_, 0, 0, 0);
    gemm_nt<128, 256, 2, 4, 2><<<dim3(E_ / 256, BS_ / 128, 1), blk, 0, stream>>>(
        Tk, wuk, ku_b, xk, BS_, E_, R_, R_, R_, E_, 0, 0, 0);
    gemm_nt<128, 256, 2, 4, 3><<<dim3(E_ / 256, BS_ / 128, 1), blk, 0, stream>>>(
        Tv, wuv, vu_b, xv, BS_, E_, R_, R_, R_, E_, 0, 0, 0);

    // 4) linear-attention contractions (small: ~3% of FLOPs)
    ksum_kernel<<<B_ * H_, blk, 0, stream>>>(xk, ksum);
    kv_kernel<<<B_ * H_, blk, 0, stream>>>(xk, xv, kvTf);
    cvt_f32_bf16<<<(B_ * H_ * D_ * D_) / 1024, blk, 0, stream>>>(
        kvTf, kvTb, B_ * H_ * D_ * D_);
    denom_kernel<<<(B_ * H_ * S_) / 256, blk, 0, stream>>>(xq, ksum, deno);

    // 5) num = qfeat @ kvT^T, batched over B*H (WMMA, K=64)
    gemm_nt<128, 64, 4, 2, 4><<<dim3(1, S_ / 128, B_ * H_), blk, 0, stream>>>(
        xq, kvTb, nullptr, numb, S_, D_, D_, D_, D_, D_,
        (long long)S_ * D_, (long long)D_ * D_, (long long)S_ * D_);

    // 6) normalize and pack to [BS, E] bf16
    combine_kernel<<<(B_ * H_ * S_ * D_) / 256, blk, 0, stream>>>(numb, deno, attn);

    // 7) output projection: out = attn @ out_w^T + out_b   (f32 to d_out)
    gemm_nt<128, 256, 2, 4, 1><<<dim3(E_ / 256, BS_ / 128, 1), blk, 0, stream>>>(
        attn, wout, out_b, out, BS_, E_, E_, E_, E_, E_, 0, 0, 0);
}
